// HybridDiNAT_s_52432960749586
// MI455X (gfx1250) — matrix operand order repaired and checked by reading, not compiled, not run on hardware
//
#include <hip/hip_runtime.h>
#include <hip/hip_bf16.h>
#include <math.h>

typedef __bf16 bf16_t;
typedef __attribute__((ext_vector_type(16))) __bf16 v16bf;
typedef __attribute__((ext_vector_type(8)))  __bf16 v8bf;
typedef __attribute__((ext_vector_type(8)))  float  v8f;

union Frag16 { v16bf v; v8bf h[2]; };

static __device__ __forceinline__ bf16_t f2bf(float f) {
  unsigned u = __builtin_bit_cast(unsigned, f);
  unsigned r = (u + 0x7FFFu + ((u >> 16) & 1u)) >> 16;
  unsigned short s = (unsigned short)r;
  return __builtin_bit_cast(bf16_t, s);
}

// two contiguous 16B LDS chunks -> one 16-element bf16 fragment
static __device__ __forceinline__ v16bf ld_frag(const bf16_t* p0, const bf16_t* p1) {
  Frag16 u;
  u.h[0] = *(const v8bf*)p0;
  u.h[1] = *(const v8bf*)p1;
  return u.v;
}

// convert 8 floats -> 8 bf16, one 16B LDS store
static __device__ __forceinline__ void st8(bf16_t* dst, float4 a, float4 b) {
  v8bf t;
  t[0] = f2bf(a.x); t[1] = f2bf(a.y); t[2] = f2bf(a.z); t[3] = f2bf(a.w);
  t[4] = f2bf(b.x); t[5] = f2bf(b.y); t[6] = f2bf(b.z); t[7] = f2bf(b.w);
  *(v8bf*)dst = t;
}

static __device__ __forceinline__ float gelu_f(float x) {
  return 0.5f * x * (1.f + erff(x * 0.7071067811865475f));
}

// ---------------------------------------------------------------------------
// Generic bf16-WMMA GEMM: C = act(A @ op(B) + bias) + resid
// A: [M,K] f32 (lda). B: transB? [N,K] (ldb) : [K,N] (ldb). C: [M,N] f32 (ldc)
// Batched via blockIdx.z with element strides sA/sB/sC (resid uses sC).
// LDS B tile is stored transposed [n][k] so wmma fragments are contiguous.
// ---------------------------------------------------------------------------
struct GemmArgs {
  const float* A; const float* Bm; float* C;
  const float* bias; const float* resid;
  long long sA, sB, sC;
  int M, N, K;
  int lda, ldb, ldc;
  int transB, act;
};

constexpr int GBM = 64, GBN = 64, GBK = 32;

__global__ __launch_bounds__(128) void gemm_wmma_kernel(GemmArgs g) {
  __shared__ bf16_t As[GBM][GBK + 8];    // row pitch 80B (16B aligned)
  __shared__ bf16_t BsT[GBN][GBK + 8];   // transposed: [n][k]
  const int tid  = threadIdx.x;
  const int wv   = tid >> 5;
  const int lane = tid & 31;
  const int hf   = lane >> 4;
  const int l16  = lane & 15;
  const int bm0  = blockIdx.y * GBM;
  const int bn0  = blockIdx.x * GBN;
  const long long zb = blockIdx.z;
  const float* A  = g.A  + zb * g.sA;
  const float* Bp = g.Bm + zb * g.sB;
  float*       C  = g.C  + zb * g.sC;

  v8f acc[4];
#pragma unroll
  for (int t = 0; t < 4; t++)
#pragma unroll
    for (int j = 0; j < 8; j++) acc[t][j] = 0.f;

  for (int kt = 0; kt < g.K; kt += GBK) {
    { // stage A tile (f32 -> bf16): 2 threads per row, 16 elems each
      int r = tid >> 1, c0 = (tid & 1) * 16;
      int gm = bm0 + r;
      const float* ap = A + (long long)gm * g.lda + kt + c0;
      if (gm < g.M && (g.lda & 3) == 0 && kt + c0 + 16 <= g.K) {
        const float4* pf = (const float4*)ap;
        float4 f0 = pf[0], f1 = pf[1], f2 = pf[2], f3 = pf[3];
        st8(&As[r][c0], f0, f1);
        st8(&As[r][c0 + 8], f2, f3);
        if (kt + GBK < g.K) __builtin_prefetch(ap + GBK, 0, 1);
      } else {
#pragma unroll
        for (int i = 0; i < 16; i++) {
          int gk = kt + c0 + i;
          As[r][c0 + i] = f2bf((gm < g.M && gk < g.K) ? ap[i] : 0.f);
        }
      }
    }
    if (g.transB) { // B is [N,K]: rows are contiguous in K -> vector path
      int n = tid >> 1, k0 = (tid & 1) * 16;
      int gn = bn0 + n;
      const float* bp = Bp + (long long)gn * g.ldb + kt + k0;
      if (gn < g.N && (g.ldb & 3) == 0 && kt + k0 + 16 <= g.K) {
        const float4* pf = (const float4*)bp;
        float4 f0 = pf[0], f1 = pf[1], f2 = pf[2], f3 = pf[3];
        st8(&BsT[n][k0], f0, f1);
        st8(&BsT[n][k0 + 8], f2, f3);
        if (kt + GBK < g.K) __builtin_prefetch(bp + GBK, 0, 1);
      } else {
#pragma unroll
        for (int i = 0; i < 16; i++) {
          int gk = kt + k0 + i;
          BsT[n][k0 + i] = f2bf((gn < g.N && gk < g.K) ? bp[i] : 0.f);
        }
      }
    } else {        // B is [K,N]: contiguous in N, scatter into BsT
      int k = tid >> 2, n0 = (tid & 3) * 16;
      int gk = kt + k;
      const float* bp = Bp + (long long)gk * g.ldb + bn0 + n0;
      if (gk < g.K && (g.ldb & 3) == 0 && bn0 + n0 + 16 <= g.N) {
        const float4* pf = (const float4*)bp;
        float4 f0 = pf[0], f1 = pf[1], f2 = pf[2], f3 = pf[3];
        BsT[n0 +  0][k] = f2bf(f0.x); BsT[n0 +  1][k] = f2bf(f0.y);
        BsT[n0 +  2][k] = f2bf(f0.z); BsT[n0 +  3][k] = f2bf(f0.w);
        BsT[n0 +  4][k] = f2bf(f1.x); BsT[n0 +  5][k] = f2bf(f1.y);
        BsT[n0 +  6][k] = f2bf(f1.z); BsT[n0 +  7][k] = f2bf(f1.w);
        BsT[n0 +  8][k] = f2bf(f2.x); BsT[n0 +  9][k] = f2bf(f2.y);
        BsT[n0 + 10][k] = f2bf(f2.z); BsT[n0 + 11][k] = f2bf(f2.w);
        BsT[n0 + 12][k] = f2bf(f3.x); BsT[n0 + 13][k] = f2bf(f3.y);
        BsT[n0 + 14][k] = f2bf(f3.z); BsT[n0 + 15][k] = f2bf(f3.w);
        if (gk + GBK < g.K) __builtin_prefetch(Bp + (long long)(gk + GBK) * g.ldb + bn0 + n0, 0, 1);
      } else {
#pragma unroll
        for (int i = 0; i < 16; i++) {
          int gn = bn0 + n0 + i;
          BsT[n0 + i][k] = f2bf((gk < g.K && gn < g.N) ? bp[i] : 0.f);
        }
      }
    }
    __syncthreads();

    const bf16_t* arow = &As[(wv << 4) + l16][0];
    v16bf a = ld_frag(arow + (hf << 3), arow + (hf << 3) + 16);
#pragma unroll
    for (int t = 0; t < 4; t++) {
      const bf16_t* brow = &BsT[(t << 4) + l16][0];
      v16bf b = ld_frag(brow + (hf << 3), brow + (hf << 3) + 16);
      acc[t] = __builtin_amdgcn_wmma_f32_16x16x32_bf16(false, a, false, b,
                                                       (short)0, acc[t], false, false);
    }
    __syncthreads();
  }

#pragma unroll
  for (int t = 0; t < 4; t++)
#pragma unroll
    for (int j = 0; j < 8; j++) {
      int m = bm0 + (wv << 4) + j + (hf << 3);
      int n = bn0 + (t << 4) + l16;
      if (m < g.M && n < g.N) {
        float v = acc[t][j];
        if (g.bias)     v += g.bias[n];
        if (g.act == 1) v = gelu_f(v);
        if (g.resid)    v += g.resid[zb * g.sC + (long long)m * g.ldc + n];
        C[(long long)m * g.ldc + n] = v;
      }
    }
}

// ---------------------------------------------------------------------------
// Fused neighborhood (NAT) attention, d=32, KS=7. One wave per 4x4 query tile.
// Union window of a 4x4 tile's clipped 7x7 windows fits in 10x10=100 keys.
// qkv: [B,H,W,3,heads,32] f32.  out: [B,H,W,heads*32] f32.  rpb: [heads,13,13].
// V is stored transposed in LDS so attn@V fragments are contiguous.
// ---------------------------------------------------------------------------
__global__ __launch_bounds__(32) void nat_attn_kernel(
    const float* __restrict__ qkv, const float* __restrict__ rpb,
    float* __restrict__ out, int Hh, int Ww, int heads, int tilesW) {
  __shared__ bf16_t qs[16][40];    // pitch 80B
  __shared__ bf16_t ks[128][40];   // pitch 80B
  __shared__ bf16_t vT[32][136];   // V^T: [d][key], pitch 272B
  __shared__ bf16_t ps[16][136];   // probs, pitch 272B
  const int lane = threadIdx.x;
  const int hf = lane >> 4, l16 = lane & 15;
  const int tile = blockIdx.x;
  const int b = blockIdx.y;
  const int h = blockIdx.z;
  const int tY = (tile / tilesW) * 4, tX = (tile % tilesW) * 4;
  const int ubI = min(max(0, tY - 3), Hh - 10);
  const int ubJ = min(max(0, tX - 3), Ww - 10);
  const int C3 = 3 * heads * 32;
  const float qscale = 0.17677669529663687f; // 32^-0.5
  const float* Qp = qkv + (long long)b * Hh * Ww * C3 + h * 32;
  const float* Kp = Qp + heads * 32;
  const float* Vp = Qp + 2 * heads * 32;

  for (int e = lane; e < 16 * 32; e += 32) {
    int m = e >> 5, dd = e & 31;
    int qy = tY + (m >> 2), qx = tX + (m & 3);
    qs[m][dd] = f2bf(Qp[((long long)qy * Ww + qx) * C3 + dd] * qscale);
  }
  for (int e = lane; e < 128 * 32; e += 32) {
    int n = e >> 5, dd = e & 31;
    float kv = 0.f, vv = 0.f;
    if (n < 100) {
      int ki = ubI + n / 10, kj = ubJ + n % 10;
      long long off = ((long long)ki * Ww + kj) * C3 + dd;
      kv = Kp[off]; vv = Vp[off];
    }
    ks[n][dd] = f2bf(kv);
    vT[dd][n] = f2bf(vv);
  }
  __syncthreads();

  // scores = q (16x32) @ k^T (32x128): 8 WMMA tiles
  v16bf a = ld_frag(&qs[l16][hf << 3], &qs[l16][(hf << 3) + 16]);
  v8f s[8];
#pragma unroll
  for (int t = 0; t < 8; t++) {
#pragma unroll
    for (int j = 0; j < 8; j++) s[t][j] = 0.f;
    const bf16_t* krow = &ks[(t << 4) + l16][0];
    v16bf bb = ld_frag(krow + (hf << 3), krow + (hf << 3) + 16);
    s[t] = __builtin_amdgcn_wmma_f32_16x16x32_bf16(false, a, false, bb,
                                                   (short)0, s[t], false, false);
  }

  // window mask + relative position bias + softmax (16-lane shfl reductions)
#pragma unroll
  for (int j = 0; j < 8; j++) {
    int m = (hf << 3) + j;
    int qy = tY + (m >> 2), qx = tX + (m & 3);
    int si = min(max(0, qy - 3), Hh - 7);
    int sj = min(max(0, qx - 3), Ww - 7);
    float sv[8];
    float mx = -1e30f;
#pragma unroll
    for (int t = 0; t < 8; t++) {
      int n = (t << 4) + l16;
      float v = -1e30f;
      if (n < 100) {
        int ki = ubI + n / 10, kj = ubJ + n % 10;
        if (ki >= si && ki < si + 7 && kj >= sj && kj < sj + 7)
          v = s[t][j] + rpb[(h * 13 + (ki - qy + 6)) * 13 + (kj - qx + 6)];
      }
      sv[t] = v;
      mx = fmaxf(mx, v);
    }
#pragma unroll
    for (int d = 1; d < 16; d <<= 1) mx = fmaxf(mx, __shfl_xor(mx, d, 32));
    float sum = 0.f;
#pragma unroll
    for (int t = 0; t < 8; t++) { float e = __expf(sv[t] - mx); sv[t] = e; sum += e; }
#pragma unroll
    for (int d = 1; d < 16; d <<= 1) sum += __shfl_xor(sum, d, 32);
    float inv = 1.f / sum;
#pragma unroll
    for (int t = 0; t < 8; t++) ps[m][(t << 4) + l16] = f2bf(sv[t] * inv);
  }
  __syncthreads();

  // out = probs (16x128) @ V (128x32): 4 K-chunks x 2 N-tiles
  v8f o[2];
#pragma unroll
  for (int nt = 0; nt < 2; nt++)
#pragma unroll
    for (int j = 0; j < 8; j++) o[nt][j] = 0.f;
#pragma unroll
  for (int kc = 0; kc < 4; kc++) {
    const bf16_t* prow = &ps[l16][kc * 32];
    v16bf a2 = ld_frag(prow + (hf << 3), prow + (hf << 3) + 16);
#pragma unroll
    for (int nt = 0; nt < 2; nt++) {
      const bf16_t* vrow = &vT[(nt << 4) + l16][kc * 32];
      v16bf b2 = ld_frag(vrow + (hf << 3), vrow + (hf << 3) + 16);
      o[nt] = __builtin_amdgcn_wmma_f32_16x16x32_bf16(false, a2, false, b2,
                                                      (short)0, o[nt], false, false);
    }
  }
  const int Cc = heads * 32;
#pragma unroll
  for (int nt = 0; nt < 2; nt++)
#pragma unroll
    for (int j = 0; j < 8; j++) {
      int m = (hf << 3) + j;
      int qy = tY + (m >> 2), qx = tX + (m & 3);
      out[((long long)(b * Hh + qy) * Ww + qx) * Cc + h * 32 + (nt << 4) + l16] = o[nt][j];
    }
}

// ---------------------------------------------------------------------------
// Glue kernels
// ---------------------------------------------------------------------------
__global__ __launch_bounds__(128) void layernorm_kernel(
    const float* __restrict__ x, const float* __restrict__ w,
    const float* __restrict__ b, float* __restrict__ y, int T, int C) {
  int t = blockIdx.x * 4 + (threadIdx.x >> 5);
  int lane = threadIdx.x & 31;
  if (t >= T) return;
  const float* xr = x + (long long)t * C;
  float s = 0.f;
  for (int c = lane; c < C; c += 32) s += xr[c];
  for (int d = 1; d < 32; d <<= 1) s += __shfl_xor(s, d, 32);
  float mu = s / C;
  float v = 0.f;
  for (int c = lane; c < C; c += 32) { float dd = xr[c] - mu; v += dd * dd; }
  for (int d = 1; d < 32; d <<= 1) v += __shfl_xor(v, d, 32);
  float inv = rsqrtf(v / C + 1e-5f);
  float* yr = y + (long long)t * C;
  for (int c = lane; c < C; c += 32) yr[c] = (xr[c] - mu) * inv * w[c] + b[c];
}

__global__ __launch_bounds__(128) void softmax_kernel(
    float* __restrict__ s, const float* __restrict__ size, int B, int heads, int N) {
  long long rows = (long long)B * heads * N;
  long long row = (long long)blockIdx.x * 4 + (threadIdx.x >> 5);
  int lane = threadIdx.x & 31;
  if (row >= rows) return;
  int b = (int)(row / ((long long)heads * N));
  float* r = s + row * N;
  float v[8];
  int cnt = 0;
  float mx = -1e30f;
  for (int j = lane; j < N; j += 32) {
    float val = r[j] + (size ? logf(size[b * N + j]) : 0.f);
    v[cnt++] = val; mx = fmaxf(mx, val);
  }
  for (int d = 1; d < 32; d <<= 1) mx = fmaxf(mx, __shfl_xor(mx, d, 32));
  float sum = 0.f; cnt = 0;
  for (int j = lane; j < N; j += 32) { float e = __expf(v[cnt] - mx); v[cnt++] = e; sum += e; }
  for (int d = 1; d < 32; d <<= 1) sum += __shfl_xor(sum, d, 32);
  float inv = 1.f / sum; cnt = 0;
  for (int j = lane; j < N; j += 32) r[j] = v[cnt++] * inv;
}

__global__ void im2col_kernel(const float* __restrict__ x, float* __restrict__ col,
                              int B, int HW) {
  long long total = (long long)B * HW * HW * 48;
  long long i = (long long)blockIdx.x * blockDim.x + threadIdx.x;
  if (i >= total) return;
  int c48 = (int)(i % 48); long long t = i / 48;
  int j = (int)(t % HW); t /= HW;
  int ii = (int)(t % HW); int b = (int)(t / HW);
  int c = c48 >> 4, py = (c48 >> 2) & 3, px = c48 & 3;
  int img = HW * 4;
  col[i] = x[(((long long)b * 3 + c) * img + (ii * 4 + py)) * img + (j * 4 + px)];
}

__global__ void pm_gather_kernel(const float* __restrict__ x, float* __restrict__ y,
                                 int B, int H, int W, int C) {
  int H2 = H / 2, W2 = W / 2;
  long long total = (long long)B * H2 * W2 * 4 * C;
  long long i = (long long)blockIdx.x * blockDim.x + threadIdx.x;
  if (i >= total) return;
  int c = (int)(i % C); long long t = i / C;
  int q = (int)(t % 4); t /= 4;
  int j = (int)(t % W2); t /= W2;
  int ii = (int)(t % H2); int b = (int)(t / H2);
  int di = q & 1, dj = q >> 1;
  y[i] = x[(((long long)b * H + (2 * ii + di)) * W + (2 * j + dj)) * C + c];
}

__global__ void split_qkv_kernel(const float* __restrict__ qkv, float* __restrict__ q,
                                 float* __restrict__ k, float* __restrict__ v,
                                 int B, int N, int heads, int d, float qscale) {
  long long total = (long long)B * heads * N * d;
  long long i = (long long)blockIdx.x * blockDim.x + threadIdx.x;
  if (i >= total) return;
  int dd = (int)(i % d); long long t = i / d;
  int n = (int)(t % N); long long t2 = t / N;
  int h = (int)(t2 % heads); int b = (int)(t2 / heads);
  long long src = (((long long)b * N + n) * 3 * heads + h) * d + dd;
  long long hd = (long long)heads * d;
  q[i] = qkv[src] * qscale;
  k[i] = qkv[src + hd];
  v[i] = qkv[src + 2 * hd];
}

__global__ void merge_heads_kernel(const float* __restrict__ o, float* __restrict__ xa,
                                   int B, int N, int heads, int d) {
  long long total = (long long)B * heads * N * d;
  long long i = (long long)blockIdx.x * blockDim.x + threadIdx.x;
  if (i >= total) return;
  int dd = (int)(i % d); long long t = i / d;
  int n = (int)(t % N); long long t2 = t / N;
  int h = (int)(t2 % heads); int b = (int)(t2 / heads);
  xa[((long long)b * N + n) * heads * d + h * d + dd] = o[i];
}

__global__ __launch_bounds__(128) void metric_kernel(
    const float* __restrict__ k, float* __restrict__ m, int B, int N, int heads, int d) {
  int row = blockIdx.x * 4 + (threadIdx.x >> 5);
  int lane = threadIdx.x & 31;
  if (row >= B * N) return;
  int b = row / N, n = row % N;
  float vals[8];
  int cnt = 0;
  float ss = 0.f;
  for (int c = lane; c < d; c += 32) {
    float acc = 0.f;
    for (int h = 0; h < heads; h++)
      acc += k[((long long)(b * heads + h) * N + n) * d + c];
    acc /= heads;
    vals[cnt++] = acc;
    ss += acc * acc;
  }
  for (int t = 1; t < 32; t <<= 1) ss += __shfl_xor(ss, t, 32);
  float inv = rsqrtf(ss + 1e-20f);
  cnt = 0;
  for (int c = lane; c < d; c += 32) m[((long long)b * N + n) * d + c] = vals[cnt++] * inv;
}

__global__ __launch_bounds__(128) void argmax_kernel(
    const float* __restrict__ s, float* __restrict__ nmax, int* __restrict__ nidx,
    int B, int Na, int Nb) {
  int row = blockIdx.x * 4 + (threadIdx.x >> 5);
  int lane = threadIdx.x & 31;
  if (row >= B * Na) return;
  const float* r = s + (long long)row * Nb;
  float bm = -1e30f; int bi = 0x7FFFFFFF;
  for (int j = lane; j < Nb; j += 32) {
    float v = r[j];
    if (v > bm || (v == bm && j < bi)) { bm = v; bi = j; }
  }
  for (int t = 1; t < 32; t <<= 1) {
    float om = __shfl_xor(bm, t, 32);
    int oi = __shfl_xor(bi, t, 32);
    if (om > bm || (om == bm && oi < bi)) { bm = om; bi = oi; }
  }
  if (lane == 0) { nmax[row] = bm; nidx[row] = bi; }
}

__global__ void sort_kernel(const float* __restrict__ nmax, const int* __restrict__ nidx,
                            int* __restrict__ unmI, int* __restrict__ srcI,
                            int* __restrict__ dstI, int Na, int r) {
  if (threadIdx.x != 0) return;
  int b = blockIdx.x;
  int idx[128];
  for (int i = 0; i < Na; i++) idx[i] = i;
  for (int i = 1; i < Na; i++) { // stable insertion sort, descending
    int key = idx[i];
    float kv = nmax[b * Na + key];
    int j = i - 1;
    while (j >= 0 && nmax[b * Na + idx[j]] < kv) { idx[j + 1] = idx[j]; j--; }
    idx[j + 1] = key;
  }
  for (int t = 0; t < r; t++) {
    srcI[b * 32 + t] = idx[t];
    dstI[b * 32 + t] = nidx[b * Na + idx[t]];
  }
  for (int u = 0; u < Na - r; u++) unmI[b * 128 + u] = idx[r + u];
}

__global__ void merge_gather_kernel(const float* __restrict__ x, const float* __restrict__ size,
                                    float* __restrict__ xm, float* __restrict__ sizem,
                                    const int* __restrict__ unmI, int B, int N, int C,
                                    int Na, int Nb, int r) {
  int Np = Na - r + Nb;
  long long total = (long long)B * Np * C;
  long long i = (long long)blockIdx.x * blockDim.x + threadIdx.x;
  if (i >= total) return;
  int c = (int)(i % C); long long t = i / C;
  int row = (int)(t % Np); int b = (int)(t / Np);
  int srow;
  if (row < Na - r) srow = 2 * unmI[b * 128 + row];
  else              srow = 2 * (row - (Na - r)) + 1;
  float sz = size ? size[b * N + srow] : 1.f;
  xm[i] = x[((long long)b * N + srow) * C + c] * sz;
  if (c == 0) sizem[b * Np + row] = sz;
}

__global__ void merge_scatter_kernel(const float* __restrict__ x, const float* __restrict__ size,
                                     float* __restrict__ xm, float* __restrict__ sizem,
                                     const int* __restrict__ srcI, const int* __restrict__ dstI,
                                     int B, int N, int C, int Na, int r, int Np) {
  long long total = (long long)B * r * C;
  long long i = (long long)blockIdx.x * blockDim.x + threadIdx.x;
  if (i >= total) return;
  int c = (int)(i % C); long long t = i / C;
  int s = (int)(t % r); int b = (int)(t / r);
  int srow = 2 * srcI[b * 32 + s];
  int dpos = (Na - r) + dstI[b * 32 + s];
  float sz = size ? size[b * N + srow] : 1.f;
  atomicAdd(&xm[((long long)b * Np + dpos) * C + c],
            x[((long long)b * N + srow) * C + c] * sz);
  if (c == 0) atomicAdd(&sizem[b * Np + dpos], sz);
}

__global__ void merge_div_kernel(const float* __restrict__ xm, const float* __restrict__ sizem,
                                 float* __restrict__ xout, float* __restrict__ sizeOut,
                                 int B, int Np, int C) {
  long long total = (long long)B * Np * C;
  long long i = (long long)blockIdx.x * blockDim.x + threadIdx.x;
  if (i >= total) return;
  int c = (int)(i % C); long long t = i / C;
  int row = (int)(t % Np); int b = (int)(t / Np);
  float sz = sizem[b * Np + row];
  xout[i] = xm[i] / sz;
  if (c == 0) sizeOut[b * Np + row] = sz;
}

// ---------------------------------------------------------------------------
// Host orchestration
// ---------------------------------------------------------------------------
struct BlkW {
  const float *fc1_b, *fc1_w, *fc2_b, *fc2_w, *n1_b, *n1_w, *n2_b, *n2_w;
  const float *proj_b, *proj_w, *qkv_b, *qkv_w, *rpb;
};

extern "C" void kernel_launch(void* const* d_in, const int* in_sizes, int n_in,
                              void* d_out, int out_size, void* d_ws, size_t ws_size,
                              hipStream_t stream) {
  (void)in_sizes; (void)n_in; (void)ws_size;
  int idx = 0;
  const float* X = (const float*)d_in[idx++];
  auto nxt = [&]() { return (const float*)d_in[idx++]; };
  auto fillBlk = [&](BlkW& w, bool withRpb) {
    w.fc1_b = nxt(); w.fc1_w = nxt(); w.fc2_b = nxt(); w.fc2_w = nxt();
    w.n1_b = nxt();  w.n1_w = nxt();  w.n2_b = nxt();  w.n2_w = nxt();
    w.proj_b = nxt(); w.proj_w = nxt(); w.qkv_b = nxt(); w.qkv_w = nxt();
    w.rpb = withRpb ? nxt() : nullptr;
  };

  // params flattened: global0, global1, local0, local1, pe (alphabetical keys)
  BlkW g0b[6]; for (int i = 0; i < 6; i++) fillBlk(g0b[i], false);
  const float* up_fc1_b = nxt(); const float* up_fc1_w = nxt();
  const float* up_fc2_b = nxt(); const float* up_fc2_w = nxt();
  BlkW g1b[2]; for (int i = 0; i < 2; i++) fillBlk(g1b[i], false);
  BlkW l0b[2]; for (int i = 0; i < 2; i++) fillBlk(l0b[i], true);
  const float* l0m_nb = nxt(); const float* l0m_nw = nxt(); const float* l0m_rw = nxt();
  BlkW l1b[2]; for (int i = 0; i < 2; i++) fillBlk(l1b[i], true);
  const float* l1m_nb = nxt(); const float* l1m_nw = nxt(); const float* l1m_rw = nxt();
  const float* pe_cb = nxt(); const float* pe_cw = nxt();
  const float* pe_nb = nxt(); const float* pe_nw = nxt();

  // workspace bump allocator
  char* wp = (char*)d_ws;
  auto alloc = [&](size_t bytes) { char* p = wp; wp += (bytes + 255) & ~(size_t)255; return p; };
  float* bufA   = (float*)alloc(1204224ull * 4);
  float* bufB2  = (float*)alloc(1204224ull * 4);
  float* bufXN  = (float*)alloc(1204224ull * 4);
  float* bufQKV = (float*)alloc(3612672ull * 4);
  float* bufH   = (float*)alloc(4816896ull * 4);
  float* bufATT = (float*)alloc(1204224ull * 4);
  float* tq = (float*)alloc(301056ull * 4);
  float* tk = (float*)alloc(301056ull * 4);
  float* tv = (float*)alloc(301056ull * 4);
  float* to = (float*)alloc(301056ull * 4);
  float* tscore  = (float*)alloc(460992ull * 4);
  float* metricB = (float*)alloc(100352ull * 4);
  float* sc2   = (float*)alloc(38416ull * 4);
  float* nmaxB = (float*)alloc(4 * 128 * 4);
  int*   nidxB = (int*)alloc(4 * 128 * 4);
  int*   unmI  = (int*)alloc(4 * 128 * 4);
  int*   srcI  = (int*)alloc(4 * 32 * 4);
  int*   dstI  = (int*)alloc(4 * 32 * 4);
  float* xmB   = (float*)alloc(4ull * 196 * 384 * 4);
  float* sizemB = (float*)alloc(4 * 256 * 4);
  float* sizeB  = (float*)alloc(4 * 256 * 4);

  auto gemm = [&](const float* A, const float* Bm, float* C, const float* bias,
                  const float* resid, int M, int N, int K, int lda, int ldb, int ldc,
                  int transB, int act, int batch, long long sA, long long sB, long long sC) {
    GemmArgs ga{A, Bm, C, bias, resid, sA, sB, sC, M, N, K, lda, ldb, ldc, transB, act};
    dim3 grid((N + GBN - 1) / GBN, (M + GBM - 1) / GBM, batch);
    gemm_wmma_kernel<<<grid, dim3(128), 0, stream>>>(ga);
  };
  auto ln = [&](const float* xin, const float* wgt, const float* bgt, float* yout,
                int T, int C) {
    layernorm_kernel<<<dim3((T + 3) / 4), dim3(128), 0, stream>>>(xin, wgt, bgt, yout, T, C);
  };
  auto ew = [](long long total) { return dim3((unsigned)((total + 255) / 256)); };

  auto natBlock = [&](const BlkW& w, float*& xc, float*& xalt, int Hs, int Cs, int heads) {
    int T = 4 * Hs * Hs;
    ln(xc, w.n1_w, w.n1_b, bufXN, T, Cs);
    gemm(bufXN, w.qkv_w, bufQKV, w.qkv_b, nullptr, T, 3 * Cs, Cs, Cs, Cs, 3 * Cs, 1, 0, 1, 0, 0, 0);
    dim3 gr((Hs / 4) * (Hs / 4), 4, heads);
    nat_attn_kernel<<<gr, dim3(32), 0, stream>>>(bufQKV, w.rpb, bufATT, Hs, Hs, heads, Hs / 4);
    gemm(bufATT, w.proj_w, xalt, w.proj_b, xc, T, Cs, Cs, Cs, Cs, Cs, 1, 0, 1, 0, 0, 0);
    { float* tsw = xc; xc = xalt; xalt = tsw; }
    ln(xc, w.n2_w, w.n2_b, bufXN, T, Cs);
    gemm(bufXN, w.fc1_w, bufH, w.fc1_b, nullptr, T, 4 * Cs, Cs, Cs, Cs, 4 * Cs, 1, 1, 1, 0, 0, 0);
    gemm(bufH, w.fc2_w, xalt, w.fc2_b, xc, T, Cs, 4 * Cs, 4 * Cs, 4 * Cs, Cs, 1, 0, 1, 0, 0, 0);
    { float* tsw = xc; xc = xalt; xalt = tsw; }
  };

  auto tomeBlock = [&](const BlkW& w, float*& xc, float*& xalt, int& N, int Cs,
                       int heads, int r, bool& hasSize) {
    int T = 4 * N, d = Cs / heads, bh = 4 * heads;
    ln(xc, w.n1_w, w.n1_b, bufXN, T, Cs);
    gemm(bufXN, w.qkv_w, bufQKV, w.qkv_b, nullptr, T, 3 * Cs, Cs, Cs, Cs, 3 * Cs, 1, 0, 1, 0, 0, 0);
    long long tot = (long long)bh * N * d;
    split_qkv_kernel<<<ew(tot), dim3(256), 0, stream>>>(bufQKV, tq, tk, tv, 4, N, heads, d,
                                                        1.f / sqrtf((float)d));
    gemm(tq, tk, tscore, nullptr, nullptr, N, N, d, d, d, N, 1, 0, bh,
         (long long)N * d, (long long)N * d, (long long)N * N);
    softmax_kernel<<<dim3((unsigned)(((long long)bh * N + 3) / 4)), dim3(128), 0, stream>>>(
        tscore, hasSize ? sizeB : nullptr, 4, heads, N);
    gemm(tscore, tv, to, nullptr, nullptr, N, d, N, N, d, d, 0, 0, bh,
         (long long)N * N, (long long)N * d, (long long)N * d);
    merge_heads_kernel<<<ew(tot), dim3(256), 0, stream>>>(to, bufATT, 4, N, heads, d);
    gemm(bufATT, w.proj_w, xalt, w.proj_b, xc, T, Cs, Cs, Cs, Cs, Cs, 1, 0, 1, 0, 0, 0);
    { float* tsw = xc; xc = xalt; xalt = tsw; }

    if (r > 0) {
      metric_kernel<<<dim3((4 * N + 3) / 4), dim3(128), 0, stream>>>(tk, metricB, 4, N, heads, d);
      int Na = (N + 1) / 2, Nb = N / 2;
      gemm(metricB, metricB + d, sc2, nullptr, nullptr, Na, Nb, d, 2 * d, 2 * d, Nb, 1, 0, 4,
           (long long)N * d, (long long)N * d, (long long)Na * Nb);
      argmax_kernel<<<dim3((4 * Na + 3) / 4), dim3(128), 0, stream>>>(sc2, nmaxB, nidxB, 4, Na, Nb);
      sort_kernel<<<dim3(4), dim3(32), 0, stream>>>(nmaxB, nidxB, unmI, srcI, dstI, Na, r);
      int Np = Na - r + Nb;
      long long tg = (long long)4 * Np * Cs;
      merge_gather_kernel<<<ew(tg), dim3(256), 0, stream>>>(
          xc, hasSize ? sizeB : nullptr, xmB, sizemB, unmI, 4, N, Cs, Na, Nb, r);
      long long ts = (long long)4 * r * Cs;
      merge_scatter_kernel<<<ew(ts), dim3(256), 0, stream>>>(
          xc, hasSize ? sizeB : nullptr, xmB, sizemB, srcI, dstI, 4, N, Cs, Na, r, Np);
      merge_div_kernel<<<ew(tg), dim3(256), 0, stream>>>(xmB, sizemB, xalt, sizeB, 4, Np, Cs);
      { float* tsw = xc; xc = xalt; xalt = tsw; }
      N = Np; hasSize = true;
    }
    int T2 = 4 * N;
    ln(xc, w.n2_w, w.n2_b, bufXN, T2, Cs);
    gemm(bufXN, w.fc1_w, bufH, w.fc1_b, nullptr, T2, 4 * Cs, Cs, Cs, Cs, 4 * Cs, 1, 1, 1, 0, 0, 0);
    gemm(bufH, w.fc2_w, xalt, w.fc2_b, xc, T2, Cs, 4 * Cs, 4 * Cs, 4 * Cs, Cs, 1, 0, 1, 0, 0, 0);
    { float* tsw = xc; xc = xalt; xalt = tsw; }
  };

  // ---- patch embed: im2col + GEMM + LN ----
  im2col_kernel<<<ew(12544ll * 48), dim3(256), 0, stream>>>(X, bufH, 4, 56);
  gemm(bufH, pe_cw, bufATT, pe_cb, nullptr, 12544, 96, 48, 48, 48, 96, 1, 0, 1, 0, 0, 0);
  ln(bufATT, pe_nw, pe_nb, bufA, 12544, 96);

  float* xc = bufA;
  float* xalt = bufB2;

  // ---- local stage 0: 56x56, C=96, heads=3 ----
  for (int i = 0; i < 2; i++) natBlock(l0b[i], xc, xalt, 56, 96, 3);
  pm_gather_kernel<<<ew(4ll * 28 * 28 * 384), dim3(256), 0, stream>>>(xc, bufH, 4, 56, 56, 96);
  ln(bufH, l0m_nw, l0m_nb, bufXN, 4 * 28 * 28, 384);
  gemm(bufXN, l0m_rw, xalt, nullptr, nullptr, 4 * 28 * 28, 192, 384, 384, 384, 192, 1, 0, 1, 0, 0, 0);
  { float* tsw = xc; xc = xalt; xalt = tsw; }

  // ---- local stage 1: 28x28, C=192, heads=6 ----
  for (int i = 0; i < 2; i++) natBlock(l1b[i], xc, xalt, 28, 192, 6);
  pm_gather_kernel<<<ew(4ll * 14 * 14 * 768), dim3(256), 0, stream>>>(xc, bufH, 4, 28, 28, 192);
  ln(bufH, l1m_nw, l1m_nb, bufXN, 4 * 14 * 14, 768);
  gemm(bufXN, l1m_rw, xalt, nullptr, nullptr, 784, 384, 768, 768, 768, 384, 1, 0, 1, 0, 0, 0);
  { float* tsw = xc; xc = xalt; xalt = tsw; }

  // ---- global0: ToMe, C=384, heads=3, N 196 -> 49 ----
  int N = 196;
  bool hasSize = false;
  int rs[6];
  { int total = 196 - 49, base = total / 6, rem = total - base * 6;
    for (int i = 0; i < 6; i++) rs[i] = base + (i < rem ? 1 : 0); }
  for (int i = 0; i < 6; i++) tomeBlock(g0b[i], xc, xalt, N, 384, 3, rs[i], hasSize);

  // ---- upsample MLP 384 -> 768 (no residual) ----
  gemm(xc, up_fc1_w, bufH, up_fc1_b, nullptr, 4 * N, 384, 384, 384, 384, 384, 1, 1, 1, 0, 0, 0);
  gemm(bufH, up_fc2_w, xalt, up_fc2_b, nullptr, 4 * N, 768, 384, 384, 384, 768, 1, 0, 1, 0, 0, 0);
  { float* tsw = xc; xc = xalt; xalt = tsw; }

  // ---- global1: ToMe, C=768, heads=6, r=0 ----
  for (int i = 0; i < 2; i++) tomeBlock(g1b[i], xc, xalt, N, 768, 6, 0, hasSize);

  hipMemcpyAsync(d_out, xc, (size_t)out_size * sizeof(float),
                 hipMemcpyDeviceToDevice, stream);
}
